// GRUControls_68341519614342
// MI455X (gfx1250) — matrix operand order repaired and checked by reading, not compiled, not run on hardware
//
#include <hip/hip_runtime.h>
#include <hip/hip_fp16.h>

typedef __attribute__((ext_vector_type(16))) _Float16 v16h;
typedef __attribute__((ext_vector_type(8)))  _Float16 v8h;
typedef __attribute__((ext_vector_type(8)))  float    v8f;
typedef __attribute__((ext_vector_type(2)))  float    v2f;

#define HID   64
#define NEMBD 1024
#define ROWS_PER_WAVE 16
#define WAVES_PER_BLOCK 4
#define ROWS_PER_BLOCK (ROWS_PER_WAVE * WAVES_PER_BLOCK)

union V16 { v16h v; v8h h[2]; };

__device__ __forceinline__ float fast_sigmoid(float x) {
    // 1/(1+e^-x): v_exp + v_rcp, no fp32 divide sequence.
    float e = __expf(-x);
    return __builtin_amdgcn_rcpf(1.0f + e);
}

__device__ __forceinline__ float fast_tanh(float x) {
#if __has_builtin(__builtin_amdgcn_tanhf)
    return __builtin_amdgcn_tanhf(x);            // gfx1250 v_tanh_f32
#else
    x = fminf(fmaxf(x, -15.0f), 15.0f);          // avoid inf*0 -> NaN
    float e = __expf(2.0f * x);
    return (e - 1.0f) * __builtin_amdgcn_rcpf(e + 1.0f);
#endif
}

// One wave owns a 16-row batch tile for the whole scan.
// gh = h(16x64) @ W_hh.T(64x192): 12 N-tiles x 2 K-chunks -> 24 wmma f16 / step.
// h0 = token @ W_head.T via fp32 WMMA, token streamed through LDS with
// global_load_async_to_lds_b128 (double-buffered, coalesced 512B bursts).
__global__ __launch_bounds__(128) void gru_controls_kernel(
    const float* __restrict__ token,
    const float* __restrict__ W_head, const float* __restrict__ b_head,
    const float* __restrict__ W_ih,   const float* __restrict__ b_ih,
    const float* __restrict__ W_hh,   const float* __restrict__ b_hh,
    const float* __restrict__ W_out,  const float* __restrict__ b_out,
    const int*   __restrict__ ctrl_len_p,
    float* __restrict__ out)
{
    const int T = ctrl_len_p[0];

    // B-fragments of W_hh.T: frag f = 2*t + c; per lane 16 contiguous halves.
    __shared__ __align__(32)  _Float16 wB[24 * 32 * 16];                 // 24 KB
    __shared__ __align__(128) _Float16 hstage[WAVES_PER_BLOCK][16 * HID]; // 8 KB
    __shared__ float                   xstage[WAVES_PER_BLOCK][32];       // 0.5 KB
    __shared__ __align__(16)  float    tstage[WAVES_PER_BLOCK][2][16 * 32]; // 16 KB

    const int tid  = threadIdx.x;
    const int wave = tid >> 5;
    const int lane = tid & 31;
    const int ln   = lane & 15;
    const int hi   = lane >> 4;
    const int rowbase = blockIdx.x * ROWS_PER_BLOCK + wave * ROWS_PER_WAVE;

    // ---------------- one-time LDS fill: W_hh.T B-fragments ----------------
    for (int i = tid; i < 24 * 32 * 16; i += blockDim.x) {
        int idx = i & 15;
        int l   = (i >> 4) & 31;
        int f   = i >> 9;
        int t = f >> 1, c = f & 1;
        int n    = l & 15;
        int kloc = ((l >> 4) << 4) + idx;
        int g = t * 16 + n;
        int k = c * 32 + kloc;
        wB[i] = (_Float16)W_hh[g * HID + k];
    }
    xstage[wave][lane] = 0.0f;
    __syncthreads();

    // ---------------- per-lane invariant weights ----------------
    float wih0[12], wih1[12], bsum[8], bihn[4], bhhn[4];
#pragma unroll
    for (int t = 0; t < 12; ++t) {
        int g = t * 16 + ln;
        wih0[t] = W_ih[g * 2 + 0];
        wih1[t] = W_ih[g * 2 + 1];
    }
#pragma unroll
    for (int t = 0; t < 8; ++t) { int g = t * 16 + ln; bsum[t] = b_ih[g] + b_hh[g]; }
#pragma unroll
    for (int u = 0; u < 4; ++u) {
        int g = (8 + u) * 16 + ln;
        bihn[u] = b_ih[g]; bhhn[u] = b_hh[g];
    }
    __half2 wout2[32];
    float bo;
    {
        const int c = hi;
#pragma unroll
        for (int j = 0; j < 32; ++j)
            wout2[j] = __floats2half2_rn(W_out[c * HID + 2 * j], W_out[c * HID + 2 * j + 1]);
        bo = b_out[c];
    }

    // ---------------- h0 = token @ W_head.T + b_head (fp32 WMMA) ----------
    // token tile streamed via async global->LDS, double buffered.
    const uint32_t tbase = (uint32_t)(uintptr_t)&tstage[wave][0][0];
    const uint64_t gtok  = (uint64_t)(uintptr_t)token + (uint64_t)rowbase * NEMBD * 4u;

    auto issue_chunk = [&](int cb, int buf) {
        // 16 rows x 32 cols f32 = 2 KB = 4 x (32 lanes x 16B), fully coalesced.
#pragma unroll
        for (int q = 0; q < 4; ++q) {
            int unit = q * 32 + lane;
            int row  = unit >> 3;
            int cw   = unit & 7;
            uint64_t ga = gtok + ((uint64_t)row * NEMBD + (uint64_t)cb * 32 + cw * 4) * 4u;
            uint32_t la = tbase + (uint32_t)buf * 2048u + (uint32_t)unit * 16u;
            asm volatile("global_load_async_to_lds_b128 %0, %1, off"
                         :: "v"(la), "v"(ga) : "memory");
        }
    };

    v8f hcur[4] = {{}, {}, {}, {}};
    issue_chunk(0, 0);
    for (int cb = 0; cb < 32; ++cb) {
        const int buf = cb & 1;
        if (cb + 1 < 32) {
            issue_chunk(cb + 1, buf ^ 1);
            asm volatile("s_wait_asynccnt 4" ::: "memory");   // current chunk done
        } else {
            asm volatile("s_wait_asynccnt 0" ::: "memory");
        }
        const float* tp = &tstage[wave][buf][0];
        for (int kc = 0; kc < 8; ++kc) {
            const int klocal = kc * 4 + hi * 2;
            v2f a;
            a.x = tp[ln * 32 + klocal];
            a.y = tp[ln * 32 + klocal + 1];
            const int kg = cb * 32 + klocal;
#pragma unroll
            for (int u = 0; u < 4; ++u) {
                const float* wrow = W_head + (size_t)(u * 16 + ln) * NEMBD;
                v2f b; b.x = wrow[kg]; b.y = wrow[kg + 1];
                hcur[u] = __builtin_amdgcn_wmma_f32_16x16x4_f32(
                    false, a, false, b, (short)0, hcur[u], false, false);
            }
        }
    }
#pragma unroll
    for (int u = 0; u < 4; ++u) {
        float bh = b_head[u * 16 + ln];
#pragma unroll
        for (int j = 0; j < 8; ++j) hcur[u][j] += bh;
    }

    // stage h0 as f16 (row-major 16x64) for A-fragment reload
    _Float16* hst = hstage[wave];
#pragma unroll
    for (int u = 0; u < 4; ++u)
#pragma unroll
        for (int j = 0; j < 8; ++j) {
            int m = j + hi * 8;
            hst[m * HID + u * 16 + ln] = (_Float16)hcur[u][j];
        }

    // ---------------- scan ----------------
    const size_t outstride = (size_t)T * 2;
    for (int s = 0; s < T; ++s) {
        asm volatile("s_wait_dscnt 0" ::: "memory");   // prior hst/xstage stores visible

        // A-fragments of h (16x32 f16 layout), 16B-aligned contiguous runs.
        V16 a0, a1;
        {
            const _Float16* p = hst + ln * HID;
            a0.h[0] = *(const v8h*)(p + hi * 16);
            a0.h[1] = *(const v8h*)(p + 8 + hi * 16);
            a1.h[0] = *(const v8h*)(p + 32 + hi * 16);
            a1.h[1] = *(const v8h*)(p + 40 + hi * 16);
        }
        v2f xv[8];
        {
            const float* xp = xstage[wave];
#pragma unroll
            for (int j = 0; j < 8; ++j) {
                int m = j + hi * 8;
                xv[j].x = xp[m * 2 + 0];
                xv[j].y = xp[m * 2 + 1];
            }
        }

        auto loadB = [&](int f) -> v16h {
            V16 tmp;
            const _Float16* p = wB + ((f << 5) + lane) * 16;
            tmp.h[0] = *(const v8h*)(p);
            tmp.h[1] = *(const v8h*)(p + 8);
            return tmp.v;
        };

        for (int u = 0; u < 4; ++u) {
            v8f ar = {}, az = {}, an = {};
            v16h b;
            b = loadB(2 * u + 0);
            ar = __builtin_amdgcn_wmma_f32_16x16x32_f16(false, a0.v, false, b, (short)0, ar, false, false);
            b = loadB(2 * u + 1);
            ar = __builtin_amdgcn_wmma_f32_16x16x32_f16(false, a1.v, false, b, (short)0, ar, false, false);
            b = loadB(2 * (u + 4) + 0);
            az = __builtin_amdgcn_wmma_f32_16x16x32_f16(false, a0.v, false, b, (short)0, az, false, false);
            b = loadB(2 * (u + 4) + 1);
            az = __builtin_amdgcn_wmma_f32_16x16x32_f16(false, a1.v, false, b, (short)0, az, false, false);
            b = loadB(2 * (u + 8) + 0);
            an = __builtin_amdgcn_wmma_f32_16x16x32_f16(false, a0.v, false, b, (short)0, an, false, false);
            b = loadB(2 * (u + 8) + 1);
            an = __builtin_amdgcn_wmma_f32_16x16x32_f16(false, a1.v, false, b, (short)0, an, false, false);

            v8f hn = hcur[u];
#pragma unroll
            for (int j = 0; j < 8; ++j) {
                float gi_r = xv[j].x * wih0[u]     + xv[j].y * wih1[u]     + bsum[u];
                float gi_z = xv[j].x * wih0[u + 4] + xv[j].y * wih1[u + 4] + bsum[u + 4];
                float gi_n = xv[j].x * wih0[u + 8] + xv[j].y * wih1[u + 8] + bihn[u];
                float r = fast_sigmoid(ar[j] + gi_r);
                float z = fast_sigmoid(az[j] + gi_z);
                float n = fast_tanh(gi_n + r * (an[j] + bhhn[u]));
                float hv = (1.0f - z) * n + z * hn[j];
                hn[j] = hv;
                int m = j + hi * 8;
                hst[m * HID + u * 16 + ln] = (_Float16)hv;
            }
            hcur[u] = hn;
        }

        asm volatile("s_wait_dscnt 0" ::: "memory");   // h_new staged

        // x_new[m, c] = h_new[m,:] . W_out[c,:] + b_out[c]; lane: m=ln, c=hi
        float xacc;
        {
            const __half2* hp = (const __half2*)(hst + ln * HID);
            __half2 acc2 = __floats2half2_rn(0.f, 0.f);
#pragma unroll
            for (int j = 0; j < 32; ++j) acc2 = __hfma2(hp[j], wout2[j], acc2);
            xacc = __low2float(acc2) + __high2float(acc2) + bo;
        }
        xstage[wave][ln * 2 + hi] = xacc;
        out[(size_t)(rowbase + ln) * outstride + (size_t)s * 2 + hi] = xacc;
    }
}

extern "C" void kernel_launch(void* const* d_in, const int* in_sizes, int n_in,
                              void* d_out, int out_size, void* d_ws, size_t ws_size,
                              hipStream_t stream) {
    const float* token  = (const float*)d_in[0];
    const float* W_head = (const float*)d_in[1];
    const float* b_head = (const float*)d_in[2];
    const float* W_ih   = (const float*)d_in[3];
    const float* b_ih   = (const float*)d_in[4];
    const float* W_hh   = (const float*)d_in[5];
    const float* b_hh   = (const float*)d_in[6];
    const float* W_out  = (const float*)d_in[7];
    const float* b_out  = (const float*)d_in[8];
    const int*   ctrl   = (const int*)d_in[9];
    float* out = (float*)d_out;

    const int B = in_sizes[0] / NEMBD;
    dim3 grid(B / ROWS_PER_BLOCK);
    gru_controls_kernel<<<grid, ROWS_PER_BLOCK * 2, 0, stream>>>(
        token, W_head, b_head, W_ih, b_ih, W_hh, b_hh, W_out, b_out, ctrl, out);
}
